// CrossSeqAttentionLayer_20504173871395
// MI455X (gfx1250) — compile-verified, matrix operand order checked by reading, not run on hardware
//
#include <hip/hip_runtime.h>
#include <hip/hip_bf16.h>
#include <math.h>

// ---------------- types ----------------
typedef __bf16 bf16;
typedef bf16  bf16x8  __attribute__((ext_vector_type(8)));
typedef bf16  bf16x16 __attribute__((ext_vector_type(16)));
typedef float f32x8   __attribute__((ext_vector_type(8)));
typedef float f32x4   __attribute__((ext_vector_type(4)));
typedef unsigned int u32x4 __attribute__((ext_vector_type(4)));
typedef int  i32x4 __attribute__((ext_vector_type(4)));
typedef int  i32x8 __attribute__((ext_vector_type(8)));

#define B_  4
#define L_  2048
#define D_  1024
#define H_  16
#define DH_ 64
#define SCALE_ 0.125f   // 64^-0.5

#if __has_builtin(__builtin_amdgcn_tensor_load_to_lds)
#define USE_TDM 1
#else
#define USE_TDM 0
#endif

// float -> bf16 round-to-nearest-even
__device__ __forceinline__ bf16 f2bf(float f) {
    unsigned u = __builtin_bit_cast(unsigned, f);
    u += 0x7FFFu + ((u >> 16) & 1u);
    unsigned short h = (unsigned short)(u >> 16);
    return __builtin_bit_cast(bf16, h);
}

// ---------------- DPP16 16-lane reductions (pure VALU, no LDS) ----------------
template <int CTRL>
__device__ __forceinline__ float dpp_f(float x) {
    int r = __builtin_amdgcn_update_dpp(0, __builtin_bit_cast(int, x), CTRL, 0xf, 0xf, true);
    return __builtin_bit_cast(float, r);
}
__device__ __forceinline__ float rmax16(float x) {
    x = fmaxf(x, dpp_f<0xB1>(x));    // quad_perm xor1
    x = fmaxf(x, dpp_f<0x4E>(x));    // quad_perm xor2
    x = fmaxf(x, dpp_f<0x141>(x));   // row_half_mirror
    x = fmaxf(x, dpp_f<0x140>(x));   // row_mirror
    return x;
}
__device__ __forceinline__ float rsum16(float x) {
    x += dpp_f<0xB1>(x);
    x += dpp_f<0x4E>(x);
    x += dpp_f<0x141>(x);
    x += dpp_f<0x140>(x);
    return x;
}

// ---------------- async global -> LDS (CDNA5 ASYNCcnt path) ----------------
__device__ __forceinline__ void async_b128(void* lds_ptr, const void* gptr) {
    unsigned la = (unsigned)(uintptr_t)lds_ptr;            // LDS aperture: offset = addr[31:0]
    unsigned long long ga = (unsigned long long)(uintptr_t)gptr;
    asm volatile("global_load_async_to_lds_b128 %0, %1, off"
                 :: "v"(la), "v"(ga) : "memory");
}
__device__ __forceinline__ void wait_async_le0() { asm volatile("s_wait_asynccnt 0x0" ::: "memory"); }
__device__ __forceinline__ void wait_async_le4() { asm volatile("s_wait_asynccnt 0x4" ::: "memory"); }
__device__ __forceinline__ void wait_async_le8() { asm volatile("s_wait_asynccnt 0x8" ::: "memory"); }

#if USE_TDM
// ---------------- Tensor Data Mover: 2D bf16 tile -> LDS (TENSORcnt path) ----------------
// Builds D# groups per CDNA5 ISA sec 8.3/8.4. Pads each LDS row so a 64-elem row
// lands on a 72-elem stride (pad_interval: 2^(4+1)=32 dwords, pad_amount: 3+1=4 dwords).
__device__ __forceinline__ void tdm_load_2d(void* lds_ptr, const void* gptr,
                                            unsigned tensor_d0, unsigned tensor_d1,
                                            unsigned tile_d0, unsigned tile_d1,
                                            unsigned stride_d0,
                                            unsigned pad_i, unsigned pad_a) {
    unsigned long long ga = (unsigned long long)(uintptr_t)gptr;
    u32x4 g0;
    g0[0] = 1u;                                              // count=1 (valid user D#)
    g0[1] = (unsigned)(uintptr_t)lds_ptr;                    // lds_addr
    g0[2] = (unsigned)(ga & 0xFFFFFFFFu);                    // global_addr[31:0]
    g0[3] = (unsigned)((ga >> 32) & 0x1FFFFFFu) | (2u << 30);// global_addr[56:32] | type=2
    i32x8 g1;
    g1[0] = (int)((1u << 16) | (1u << 20) | (pad_i << 22) | (pad_a << 25)); // data_size=2B, pad_enable
    g1[1] = (int)((tensor_d0 & 0xFFFFu) << 16);              // tensor_dim0[15:0]
    g1[2] = (int)((tensor_d0 >> 16) | ((tensor_d1 & 0xFFFFu) << 16));
    g1[3] = (int)((tensor_d1 >> 16) | (tile_d0 << 16));      // tile_dim0
    g1[4] = (int)tile_d1;                                    // tile_dim1 (tile_dim2=0)
    g1[5] = (int)stride_d0;                                  // tensor_dim0_stride[31:0]
    g1[6] = 0;
    g1[7] = 0;
    i32x4 gz = {0, 0, 0, 0};
#if defined(__clang_major__) && (__clang_major__ >= 23)
    i32x8 gz8 = {0, 0, 0, 0, 0, 0, 0, 0};
    __builtin_amdgcn_tensor_load_to_lds(g0, g1, gz, gz, gz8, 0);
#else
    __builtin_amdgcn_tensor_load_to_lds(g0, g1, gz, gz, 0);
#endif
}
__device__ __forceinline__ void wait_tensor0() { __builtin_amdgcn_s_wait_tensorcnt(0); }
__device__ __forceinline__ void wait_tensor2() { __builtin_amdgcn_s_wait_tensorcnt(2); }
#endif

// 16x32 bf16 fragment load (CDNA5 16-bit A-matrix layout)
__device__ __forceinline__ bf16x16 load_frag(const bf16* p, int stride, int lane) {
    int hf = lane >> 4;
    int r  = lane & 15;
    const bf16* q = p + r * stride + hf * 8;
    bf16x8 lo = *(const bf16x8*)(q);
    bf16x8 hi = *(const bf16x8*)(q + 16);
    bf16x16 f;
#pragma unroll
    for (int i = 0; i < 8; ++i) { f[i] = lo[i]; f[i + 8] = hi[i]; }
    return f;
}

__device__ __forceinline__ f32x8 wmma_bf16(bf16x16 a, bf16x16 b, f32x8 c) {
    return __builtin_amdgcn_wmma_f32_16x16x32_bf16(false, a, false, b, (short)0, c, false, false);
}

// ---------------- prep kernels ----------------
__global__ __launch_bounds__(256) void cvt_bf16_kernel(const float* __restrict__ in,
                                                       bf16* __restrict__ out, int n) {
    int i = (blockIdx.x * 256 + threadIdx.x) * 4;
#pragma unroll
    for (int j = 0; j < 4; ++j)
        if (i + j < n) out[i + j] = f2bf(in[i + j]);
}

// beta*log(max(R,eps)) permuted into WMMA D-fragment order
__global__ __launch_bounds__(256) void prep_logr_perm(const float* __restrict__ R,
                                                      const float* __restrict__ beta,
                                                      float* __restrict__ out) {
    int kg = blockIdx.x * 256 + threadIdx.x;
    int qg = blockIdx.y;
    float b = beta[0];
    float val = b * __logf(fmaxf(R[(size_t)qg * L_ + kg], 1e-8f));
    int hf = (qg >> 3) & 1, v = qg & 7;
    int lc = kg & 15, j = (kg >> 4) & 1;
    int lane = hf * 16 + lc, e = j * 8 + v;
    out[(((size_t)(qg >> 4) * (L_ / 32) + (kg >> 5)) * 32 + lane) * 16 + e] = val;
}

// ---------------- tiled WMMA GEMM, TDM double-buffered pipeline ----------------
// C[M,N] = A[M,K] (bf16 row-major) x W.T, W bf16 row-major [N,K] (raw W = [n][k] B-tile).
// MODE 0: bf16 head-major [B,H,L,Dh]; MODE 1: f32 row-major; MODE 2: bf16 [B,H,Dh,L].
template <int MODE>
__global__ __launch_bounds__(256) void gemm_wmma(const bf16* __restrict__ A,
                                                 const bf16* __restrict__ W,
                                                 void* __restrict__ outp,
                                                 int M, int N, int K) {
    __shared__ bf16 sA[2][128 * 72];
    __shared__ bf16 sB[2][128 * 72];

    int tid = threadIdx.x, lane = tid & 31, wid = tid >> 5;
    int wm = wid & 3, wn = wid >> 2;
    int rowBase = blockIdx.y * 128, colBase = blockIdx.x * 128;
    const bf16* gA0 = A + (size_t)rowBase * K;
    const bf16* gW0 = W + (size_t)colBase * K;

    f32x8 acc[2][4];
#pragma unroll
    for (int i = 0; i < 2; ++i)
#pragma unroll
        for (int j = 0; j < 4; ++j) acc[i][j] = (f32x8)0.0f;

    const int NK = K >> 6;

#if USE_TDM
    bool leader = (wid == 0);
    if (leader) {   // one descriptor pair per tile; TDM DMAs 128x64 bf16 with HW padding
        tdm_load_2d(&sA[0][0], gA0, (unsigned)K, (unsigned)M, 64, 128, (unsigned)K, 4, 3);
        tdm_load_2d(&sB[0][0], gW0, (unsigned)K, (unsigned)N, 64, 128, (unsigned)K, 4, 3);
    }
#else
    int r = tid >> 1, hh = tid & 1;
    const bf16* gA = gA0 + (size_t)r * K + hh * 32;
    const bf16* gW = gW0 + (size_t)r * K + hh * 32;
    int lbase = r * 72 + hh * 32;
#pragma unroll
    for (int i = 0; i < 4; ++i) {
        async_b128(&sA[0][lbase + i * 8], gA + i * 8);
        async_b128(&sB[0][lbase + i * 8], gW + i * 8);
    }
#endif

    for (int kt = 0; kt < NK; ++kt) {
        int cur = kt & 1;
#if USE_TDM
        if (leader) {
            if (kt + 1 < NK) {
                tdm_load_2d(&sA[cur ^ 1][0], gA0 + (kt + 1) * 64, (unsigned)K, (unsigned)M,
                            64, 128, (unsigned)K, 4, 3);
                tdm_load_2d(&sB[cur ^ 1][0], gW0 + (kt + 1) * 64, (unsigned)K, (unsigned)N,
                            64, 128, (unsigned)K, 4, 3);
                wait_tensor2();   // tensor ops in-order: current tile's pair complete
            } else {
                wait_tensor0();
            }
        }
#else
        if (kt + 1 < NK) {
            const bf16* ga = gA + (size_t)(kt + 1) * 64;
            const bf16* gb = gW + (size_t)(kt + 1) * 64;
#pragma unroll
            for (int i = 0; i < 4; ++i) {
                async_b128(&sA[cur ^ 1][lbase + i * 8], ga + i * 8);
                async_b128(&sB[cur ^ 1][lbase + i * 8], gb + i * 8);
            }
            wait_async_le8();
        } else {
            wait_async_le0();
        }
#endif
        __syncthreads();

#pragma unroll
        for (int ks = 0; ks < 2; ++ks) {
            bf16x16 af[2], bfr[4];
#pragma unroll
            for (int fm = 0; fm < 2; ++fm)
                af[fm] = load_frag(&sA[cur][(wm * 32 + fm * 16) * 72 + ks * 32], 72, lane);
#pragma unroll
            for (int fn = 0; fn < 4; ++fn)
                bfr[fn] = load_frag(&sB[cur][(wn * 64 + fn * 16) * 72 + ks * 32], 72, lane);
#pragma unroll
            for (int fm = 0; fm < 2; ++fm)
#pragma unroll
                for (int fn = 0; fn < 4; ++fn)
                    acc[fm][fn] = wmma_bf16(af[fm], bfr[fn], acc[fm][fn]);
        }
        __syncthreads();
    }

    int hf = lane >> 4, lc = lane & 15;
#pragma unroll
    for (int fm = 0; fm < 2; ++fm)
#pragma unroll
        for (int fn = 0; fn < 4; ++fn)
#pragma unroll
            for (int v = 0; v < 8; ++v) {
                int rr = rowBase + wm * 32 + fm * 16 + v + hf * 8;
                int cc = colBase + wn * 64 + fn * 16 + lc;
                float val = acc[fm][fn][v];
                int b = rr >> 11, l = rr & (L_ - 1);
                int h = cc >> 6, d = cc & 63;
                if (MODE == 0) {
                    ((bf16*)outp)[(((size_t)(b * H_ + h)) * L_ + l) * DH_ + d] = f2bf(val);
                } else if (MODE == 2) {
                    ((bf16*)outp)[(((size_t)(b * H_ + h)) * DH_ + d) * L_ + l] = f2bf(val);
                } else {
                    ((float*)outp)[(size_t)rr * N + cc] = val;
                }
            }
}

// ---------------- flash attention with log-R bias (async double-buffered) ----------------
__global__ __launch_bounds__(128) void attn_flash(const bf16* __restrict__ Q,
                                                  const bf16* __restrict__ Km,
                                                  const bf16* __restrict__ Vt,
                                                  const float* __restrict__ logRp,
                                                  bf16* __restrict__ O) {
    __shared__ bf16 sK[2][32 * 72];    // [key][d]  (B-layout for Q@K^T)
    __shared__ bf16 sVt[2][64 * 40];   // [d][key]  (natural copy of global V^T)
    __shared__ bf16 sP[4 * 16 * 40];   // per-wave P staging

    int tid = threadIdx.x, lane = tid & 31, w = tid >> 5;
    int bid = blockIdx.x;
    int qt = bid & 31, h = (bid >> 5) & (H_ - 1), b = bid >> 9;
    int bh = b * H_ + h;
    const bf16* Qh = Q + (size_t)bh * L_ * DH_;
    const bf16* Kh = Km + (size_t)bh * L_ * DH_;
    const bf16* Vh = Vt + (size_t)bh * DH_ * L_;   // [Dh][L]
    int q0 = qt * 64 + w * 16;
    int hf = lane >> 4, lc = lane & 15;

    bf16x16 qf[2];
    qf[0] = load_frag(Qh + (size_t)q0 * DH_ + 0, DH_, lane);
    qf[1] = load_frag(Qh + (size_t)q0 * DH_ + 32, DH_, lane);

    float m[8], s[8];
    f32x8 of[4];
#pragma unroll
    for (int v = 0; v < 8; ++v) { m[v] = -1e30f; s[v] = 0.0f; }
#pragma unroll
    for (int n = 0; n < 4; ++n) of[n] = (f32x8)0.0f;

    const float* lrbase = logRp + ((size_t)(q0 >> 4) * (L_ / 32)) * 512 + lane * 16;

    int keyA = tid >> 2, segA = tid & 3;
    int dV = tid >> 1, hV = tid & 1;
    const bf16* gk0 = Kh + (size_t)keyA * DH_ + segA * 16;
    const bf16* gv0 = Vh + (size_t)dV * L_ + hV * 16;
    int lK = keyA * 72 + segA * 16;
    int lV = dV * 40 + hV * 16;

    async_b128(&sK[0][lK],      gk0);
    async_b128(&sK[0][lK + 8],  gk0 + 8);
    async_b128(&sVt[0][lV],     gv0);
    async_b128(&sVt[0][lV + 8], gv0 + 8);

    for (int kt = 0; kt < L_; kt += 32) {
        int cur = (kt >> 5) & 1;
        if (kt + 32 < L_) {
            const bf16* gk = gk0 + (size_t)(kt + 32) * DH_;
            const bf16* gv = gv0 + (kt + 32);
            async_b128(&sK[cur ^ 1][lK],      gk);
            async_b128(&sK[cur ^ 1][lK + 8],  gk + 8);
            async_b128(&sVt[cur ^ 1][lV],     gv);
            async_b128(&sVt[cur ^ 1][lV + 8], gv + 8);
        }

        const float* lr = lrbase + (size_t)(kt >> 5) * 512;
        f32x4 c0 = *(const f32x4*)(lr + 0);
        f32x4 c1 = *(const f32x4*)(lr + 4);
        f32x4 c2 = *(const f32x4*)(lr + 8);
        f32x4 c3 = *(const f32x4*)(lr + 12);
        if (kt + 32 < L_) __builtin_prefetch(lr + 512, 0, 1);

        if (kt + 32 < L_) wait_async_le4(); else wait_async_le0();
        __syncthreads();

        f32x8 sc[2];
        sc[0] = (f32x8)0.0f; sc[1] = (f32x8)0.0f;
#pragma unroll
        for (int j = 0; j < 2; ++j)
#pragma unroll
            for (int kd = 0; kd < 2; ++kd)
                sc[j] = wmma_bf16(qf[kd],
                                  load_frag(&sK[cur][(j * 16) * 72 + kd * 32], 72, lane), sc[j]);

        float xv[2][8];
#pragma unroll
        for (int v = 0; v < 4; ++v) {
            xv[0][v]     = sc[0][v]     * SCALE_ + c0[v];
            xv[0][v + 4] = sc[0][v + 4] * SCALE_ + c1[v];
            xv[1][v]     = sc[1][v]     * SCALE_ + c2[v];
            xv[1][v + 4] = sc[1][v + 4] * SCALE_ + c3[v];
        }

        float mn[8], corr[8];
#pragma unroll
        for (int v = 0; v < 8; ++v) {
            float rm = rmax16(fmaxf(xv[0][v], xv[1][v]));
            mn[v] = fmaxf(m[v], rm);
            corr[v] = __expf(m[v] - mn[v]);
            m[v] = mn[v];
        }
#pragma unroll
        for (int j = 0; j < 2; ++j)
#pragma unroll
            for (int v = 0; v < 8; ++v) xv[j][v] = __expf(xv[j][v] - mn[v]);
#pragma unroll
        for (int v = 0; v < 8; ++v)
            s[v] = s[v] * corr[v] + rsum16(xv[0][v] + xv[1][v]);
#pragma unroll
        for (int n = 0; n < 4; ++n)
#pragma unroll
            for (int v = 0; v < 8; ++v) of[n][v] *= corr[v];

        bf16* myP = &sP[w * 16 * 40];
#pragma unroll
        for (int j = 0; j < 2; ++j)
#pragma unroll
            for (int v = 0; v < 8; ++v)
                myP[(v + hf * 8) * 40 + j * 16 + lc] = f2bf(xv[j][v]);

        bf16x16 pf = load_frag(myP, 40, lane);
#pragma unroll
        for (int n = 0; n < 4; ++n)
            of[n] = wmma_bf16(pf, load_frag(&sVt[cur][(n * 16) * 40], 40, lane), of[n]);
        __syncthreads();
    }

#pragma unroll
    for (int v = 0; v < 8; ++v) s[v] = 1.0f / s[v];
#pragma unroll
    for (int n = 0; n < 4; ++n)
#pragma unroll
        for (int v = 0; v < 8; ++v) {
            int qg = q0 + v + hf * 8;
            int d = n * 16 + lc;
            O[((size_t)(b * L_ + qg)) * D_ + h * DH_ + d] = f2bf(of[n][v] * s[v]);
        }
}

// ---------------- residual + LayerNorm ----------------
__global__ __launch_bounds__(256) void resid_ln(const float* __restrict__ Y,
                                                const float* __restrict__ tokens,
                                                const float* __restrict__ gamma,
                                                const float* __restrict__ bias,
                                                float* __restrict__ out) {
    int row = blockIdx.x, tid = threadIdx.x;
    const float* y = Y + (size_t)row * D_;
    const float* t = tokens + (size_t)row * D_;
    float x[4], sum = 0.f, sq = 0.f;
#pragma unroll
    for (int i = 0; i < 4; ++i) {
        int c = tid + i * 256;
        x[i] = y[c] + t[c];
        sum += x[i];
        sq  += x[i] * x[i];
    }
    sum = rsum16(sum); sq = rsum16(sq);
    __shared__ float red[32];
    int wid = tid >> 5, lane = tid & 31;
    if ((lane & 15) == 0) { red[wid * 2 + (lane >> 4)] = sum; red[16 + wid * 2 + (lane >> 4)] = sq; }
    __syncthreads();
    sum = 0.f; sq = 0.f;
#pragma unroll
    for (int i = 0; i < 16; ++i) { sum += red[i]; sq += red[16 + i]; }
    float mu = sum * (1.0f / D_);
    float var = sq * (1.0f / D_) - mu * mu;
    float rr = rsqrtf(var + 1e-5f);
#pragma unroll
    for (int i = 0; i < 4; ++i) {
        int c = tid + i * 256;
        out[(size_t)row * D_ + c] = (x[i] - mu) * rr * gamma[c] + bias[c];
    }
}

// ---------------- host launcher ----------------
extern "C" void kernel_launch(void* const* d_in, const int* in_sizes, int n_in,
                              void* d_out, int out_size, void* d_ws, size_t ws_size,
                              hipStream_t stream) {
    const float* tokens = (const float*)d_in[0];
    const float* R      = (const float*)d_in[1];
    const float* Wq     = (const float*)d_in[2];
    const float* Wk     = (const float*)d_in[3];
    const float* Wv     = (const float*)d_in[4];
    const float* Wp     = (const float*)d_in[5];
    const float* beta   = (const float*)d_in[6];
    const float* gamma  = (const float*)d_in[7];
    const float* bias   = (const float*)d_in[8];
    float* out = (float*)d_out;

    const int M = B_ * L_;   // 8192
    char* ws = (char*)d_ws;
    size_t off = 0;
    bf16*  Xbf   = (bf16*)(ws + off);  off += (size_t)M * D_ * 2;
    bf16*  Wb    = (bf16*)(ws + off);  off += (size_t)4 * D_ * D_ * 2;
    bf16*  Qb    = (bf16*)(ws + off);  off += (size_t)M * D_ * 2;
    bf16*  Kb    = (bf16*)(ws + off);  off += (size_t)M * D_ * 2;
    bf16*  Vtb   = (bf16*)(ws + off);  off += (size_t)M * D_ * 2;
    float* logRp = (float*)(ws + off); off += (size_t)L_ * L_ * 4;
    bf16*  Ob    = (bf16*)(ws + off);  off += (size_t)M * D_ * 2;
    float* Y     = (float*)(ws + off); off += (size_t)M * D_ * 4;

    cvt_bf16_kernel<<<(M * D_) / 1024, 256, 0, stream>>>(tokens, Xbf, M * D_);
    cvt_bf16_kernel<<<(D_ * D_) / 1024, 256, 0, stream>>>(Wq, Wb + 0 * (size_t)D_ * D_, D_ * D_);
    cvt_bf16_kernel<<<(D_ * D_) / 1024, 256, 0, stream>>>(Wk, Wb + 1 * (size_t)D_ * D_, D_ * D_);
    cvt_bf16_kernel<<<(D_ * D_) / 1024, 256, 0, stream>>>(Wv, Wb + 2 * (size_t)D_ * D_, D_ * D_);
    cvt_bf16_kernel<<<(D_ * D_) / 1024, 256, 0, stream>>>(Wp, Wb + 3 * (size_t)D_ * D_, D_ * D_);
    prep_logr_perm<<<dim3(L_ / 256, L_), 256, 0, stream>>>(R, beta, logRp);

    dim3 ggrid(D_ / 128, M / 128);
    gemm_wmma<0><<<ggrid, 256, 0, stream>>>(Xbf, Wb + 0 * (size_t)D_ * D_, Qb, M, D_, D_);
    gemm_wmma<0><<<ggrid, 256, 0, stream>>>(Xbf, Wb + 1 * (size_t)D_ * D_, Kb, M, D_, D_);
    gemm_wmma<2><<<ggrid, 256, 0, stream>>>(Xbf, Wb + 2 * (size_t)D_ * D_, Vtb, M, D_, D_);

    attn_flash<<<B_ * H_ * (L_ / 64), 128, 0, stream>>>(Qb, Kb, Vtb, logRp, Ob);

    gemm_wmma<1><<<ggrid, 256, 0, stream>>>(Ob, Wb + 3 * (size_t)D_ * D_, Y, M, D_, D_);

    resid_ln<<<M, 256, 0, stream>>>(Y, tokens, gamma, bias, out);
}
